// RNN_6167573037204
// MI455X (gfx1250) — compile-verified, matrix operand order checked by reading, not run on hardware
//
#include <hip/hip_runtime.h>

typedef __attribute__((ext_vector_type(16))) _Float16 v16h;
typedef __attribute__((ext_vector_type(8)))  float    v8f;

// may_alias views so f16 stores / u32 loads of the h buffer keep program order
// (single-wave workgroup: LDS ops from one wave execute in order, so ordinary
//  dependence order is all we need -- no barrier drains on the critical path)
typedef _Float16     f16a __attribute__((may_alias));
typedef unsigned int u32a __attribute__((may_alias));
typedef float        f32a __attribute__((may_alias));

#define HDIM 24
#define GSTR 100    // gate LDS row stride (dwords); halves land 32 banks apart -> conflict-free
#define HROW 34     // h LDS row stride in f16 halfwords (= 17 dwords, coprime with 64 banks)
#define HDW  17     // h LDS row stride in dwords
#define ZSTR 27     // z LDS row stride (dwords)

__device__ __forceinline__ float fast_tanh(float v) {
#if __has_builtin(__builtin_amdgcn_tanhf)
    return __builtin_amdgcn_tanhf(v);           // v_tanh_f32
#elif __has_builtin(__builtin_amdgcn_tanh_f32)
    return __builtin_amdgcn_tanh_f32(v);
#else
    return 2.0f / (1.0f + __expf(-2.0f * v)) - 1.0f;
#endif
}
__device__ __forceinline__ float fast_sigmoid(float v) {
    return 0.5f * fast_tanh(0.5f * v) + 0.5f;   // one trans op + fma
}

__device__ __forceinline__ void sched_fence() {
#if __has_builtin(__builtin_amdgcn_sched_barrier)
    __builtin_amdgcn_sched_barrier(0);          // nothing crosses: pins live ranges
#endif
}

union AView { v16h v; unsigned int u[8]; };

// Rebuild the f16 A matrix (16x32, augmented [h | 1 | x | 0..]) from packed-f16 LDS.
__device__ __forceinline__ v16h rebuild_A(const u32a* lds_hw, int col, int half) {
    AView a;
#pragma unroll
    for (int v = 0; v < 8; ++v) {
        // slots (2v,2v+1) hold K pair; dword index = Kbase/2
        int sub = (v < 4) ? (v + 4 * half) : (v + 4 + 4 * half);
        a.u[v] = lds_hw[col * HDW + sub];
    }
    return a.v;
}

// One LSTM timestep for a 16-row batch tile owned by a single wave32.
// Bias and x*W_ih ride inside the WMMA via augmented K rows 24 (ones) and 25 (x).
__device__ __forceinline__ void lstm_step(float* creg, const v16h* Bm, v16h& A,
                                          float xnext, f32a* lds_g,
                                          f16a* lds_h, const u32a* lds_hw,
                                          int col, int half, int lane)
{
    const v8f zeroC = {};

    // gates[16x96] = [h | 1 | x] @ [W_hh^T ; bias ; W_ih]
    v8f D[6];
#pragma unroll
    for (int j = 0; j < 6; ++j)
        D[j] = __builtin_amdgcn_wmma_f32_16x16x32_f16(
            false, A, false, Bm[j], (short)0, zeroC, false, false);
    sched_fence();   // all 6 accumulators live here -> 6 independent, pipelined WMMAs

#pragma unroll
    for (int j = 0; j < 6; ++j)
#pragma unroll
        for (int r = 0; r < 8; ++r)
            lds_g[(r + 8 * half) * GSTR + 16 * j + col] = D[j][r];

    // activations + state update: 16 rows x 24 units = 384 cells, 12 per lane
    // (loads may-alias the stores above -> issued after them; LDS is in-order per wave)
#pragma unroll
    for (int q = 0; q < 12; ++q) {
        int flat = lane + 32 * q;                       // 0..383
        int b = flat / HDIM, k = flat % HDIM;
        float gi = fast_sigmoid(lds_g[b * GSTR +  0 + k]);
        float gf = fast_sigmoid(lds_g[b * GSTR + 24 + k]);
        float gg = fast_tanh   (lds_g[b * GSTR + 48 + k]);
        float go = fast_sigmoid(lds_g[b * GSTR + 72 + k]);
        float c  = gf * creg[q] + gi * gg;
        creg[q]  = c;
        lds_h[b * HROW + k] = (_Float16)(go * fast_tanh(c));
    }
    if (half)                                           // row col's x for the NEXT step
        lds_h[col * HROW + 25] = (_Float16)xnext;

    A = rebuild_A(lds_hw, col, half);                   // 8 uniform b32 loads, no converts
}

__global__ __launch_bounds__(32)
void lstm_wmma_kernel(const float* __restrict__ x,      // [B*T]
                      const float* __restrict__ h0,     // [B*H]
                      const float* __restrict__ c0,     // [B*H]
                      const float* __restrict__ W_ih,   // [96]
                      const float* __restrict__ W_hh,   // [96*24] row-major
                      const float* __restrict__ b_ih,   // [96]
                      const float* __restrict__ b_hh,   // [96]
                      const float* __restrict__ W1,     // [24*24]
                      const float* __restrict__ b1,     // [24]
                      const float* __restrict__ W2,     // [24]
                      const float* __restrict__ b2,     // [1]
                      float* __restrict__ out,          // [B]
                      int Tlen)
{
    __shared__ float        lds_g_[16 * GSTR];
    __shared__ unsigned int lds_hw_[16 * HDW];          // h as packed f16 pairs (dword view)
    __shared__ float        lds_z_[16 * ZSTR];

    f32a*       lds_g  = (f32a*)lds_g_;
    f16a*       lds_h  = (f16a*)lds_hw_;
    const u32a* lds_hw = (const u32a*)lds_hw_;
    f32a*       lds_z  = (f32a*)lds_z_;

    const int lane = threadIdx.x & 31;
    const int col  = lane & 15;
    const int half = lane >> 4;
    const int b0   = blockIdx.x * 16;

    // ---- one-time setup: augmented W as 6 f16 B-matrix register tiles ----
    // Rows K<24: W_hh^T. Row K=24: bias (b_ih+b_hh). Row K=25: W_ih. Rows 26..31: 0.
    v16h Bm[6];
#pragma unroll
    for (int j = 0; j < 6; ++j) {
        const int g = 16 * j + col;                     // output gate column
#pragma unroll
        for (int i = 0; i < 16; ++i) {
            int K = ((i < 8) ? i : i + 8) + 8 * half;   // K index for this slot
            float w;
            if (K < HDIM)           w = W_hh[g * HDIM + K];
            else if (K == HDIM)     w = b_ih[g] + b_hh[g];
            else if (K == HDIM + 1) w = W_ih[g];
            else                    w = 0.0f;
            Bm[j][i] = (_Float16)w;
        }
    }

    // ---- initial state: c into registers, h0 (f16) + augmentation into LDS ----
    float creg[12];
#pragma unroll
    for (int q = 0; q < 12; ++q) {
        int flat = lane + 32 * q;
        int b = flat / HDIM, k = flat % HDIM;
        creg[q] = c0[(b0 + b) * HDIM + k];
        lds_h[b * HROW + k] = (_Float16)h0[(b0 + b) * HDIM + k];
    }
    if (lane < 16) {                                    // row `lane`: constant aug slots
        lds_h[lane * HROW + 24] = (_Float16)1.0f;       // ones row (bias)
#pragma unroll
        for (int k = 26; k < HROW; ++k)
            lds_h[lane * HROW + k] = (_Float16)0.0f;    // K=26..31 padding
    }

    // every lane streams row `col` of x; the half=1 copy feeds A slot K=25
    const float* xrow = x + (size_t)(b0 + col) * (size_t)Tlen;
    float x0 = xrow[0];
    if (half) lds_h[col * HROW + 25] = (_Float16)x0;    // x for t=0
    __syncthreads();

    v16h A = rebuild_A(lds_hw, col, half);

    // ---- scan over time; step t also stages x(t+1) ----
    if ((Tlen & 3) == 0) {
        float4 cur = *(const float4*)(xrow);            // t0..t0+3
        for (int t0 = 0; t0 < Tlen; t0 += 4) {
            float4 nxt;
            if (t0 + 4 < Tlen) nxt = *(const float4*)(xrow + t0 + 4);
            else { nxt.x = 0.f; nxt.y = 0.f; nxt.z = 0.f; nxt.w = 0.f; }
            lstm_step(creg, Bm, A, cur.y, lds_g, lds_h, lds_hw, col, half, lane);
            lstm_step(creg, Bm, A, cur.z, lds_g, lds_h, lds_hw, col, half, lane);
            lstm_step(creg, Bm, A, cur.w, lds_g, lds_h, lds_hw, col, half, lane);
            lstm_step(creg, Bm, A, nxt.x, lds_g, lds_h, lds_hw, col, half, lane);
            cur = nxt;
        }
    } else {                                            // generic fallback
        for (int t = 0; t < Tlen; ++t) {
            float xn = (t + 1 < Tlen) ? xrow[t + 1] : 0.0f;
            lstm_step(creg, Bm, A, xn, lds_g, lds_h, lds_hw, col, half, lane);
        }
    }
    __syncthreads();

    // ---- fused MLP head on LDS-resident h_n (f16) ----
    // z[b,j] = relu(h[b,:] . W1[j,:] + b1[j])  : 384 cells, 12 per lane
#pragma unroll
    for (int q = 0; q < 12; ++q) {
        int flat = lane + 32 * q;
        int b = flat / HDIM, j = flat % HDIM;
        float acc = b1[j];
#pragma unroll
        for (int k = 0; k < HDIM; ++k)
            acc += (float)lds_h[b * HROW + k] * W1[j * HDIM + k];
        lds_z[b * ZSTR + j] = fmaxf(acc, 0.0f);
    }
    __syncthreads();

    // out[b] = relu(z[b,:] . W2 + b2)
    if (lane < 16) {
        float acc = b2[0];
#pragma unroll
        for (int j = 0; j < HDIM; ++j)
            acc += lds_z[lane * ZSTR + j] * W2[j];
        out[b0 + lane] = fmaxf(acc, 0.0f);
    }
}

extern "C" void kernel_launch(void* const* d_in, const int* in_sizes, int n_in,
                              void* d_out, int out_size, void* d_ws, size_t ws_size,
                              hipStream_t stream) {
    const float* x    = (const float*)d_in[0];
    const float* h0   = (const float*)d_in[1];
    const float* c0   = (const float*)d_in[2];
    // d_in[3] = y : unused by the forward reference
    const float* W_ih = (const float*)d_in[4];
    const float* W_hh = (const float*)d_in[5];
    const float* b_ih = (const float*)d_in[6];
    const float* b_hh = (const float*)d_in[7];
    const float* W1   = (const float*)d_in[8];
    const float* b1   = (const float*)d_in[9];
    const float* W2   = (const float*)d_in[10];
    const float* b2   = (const float*)d_in[11];

    const int B    = in_sizes[1] / HDIM;     // h_state: [1,B,H]
    const int Tlen = in_sizes[0] / B;        // x: [B,T,1]

    dim3 grid(B / 16), block(32);
    hipLaunchKernelGGL(lstm_wmma_kernel, grid, block, 0, stream,
                       x, h0, c0, W_ih, W_hh, b_ih, b_hh, W1, b1, W2, b2,
                       (float*)d_out, Tlen);
}
